// VSSBlock_54451595378865
// MI455X (gfx1250) — matrix operand order repaired and checked by reading, not compiled, not run on hardware
//
#include <hip/hip_runtime.h>
#include <hip/hip_bf16.h>

typedef __attribute__((ext_vector_type(16))) __bf16          v16bf;
typedef __attribute__((ext_vector_type(16))) unsigned short  v16u;
typedef __attribute__((ext_vector_type(8)))  float           v8f;

__device__ __forceinline__ unsigned short f2bf(float f) {
  unsigned int u = __builtin_bit_cast(unsigned int, f);
  u += 0x7FFFu + ((u >> 16) & 1u);            // round-to-nearest-even
  return (unsigned short)(u >> 16);
}

// ---------------------------------------------------------------------------
// Weight packing with PERMUTED K ordering:  K' = tap*I + ci  (tap = kh*KW+kw).
// 32-wide K'-chunks then lie inside one tap (I=96 -> 3 chunks per tap), so the
// GEMM's B gather needs no per-lane divides and is one contiguous LDS load.
// Fragment order uses contiguous k per lane half (same order as the B loads):
//   dst[i], i = ((kc*6 + mt)*32 + lane)*16 + e
//   m  = mt*16 + (lane&15)
//   K' = kc*32 + ((lane&16)?16:0) + e
//   ci = K' % I, tap = K' / I, src = w[(m*I+ci)*KHW + tap]
// ---------------------------------------------------------------------------
__global__ __launch_bounds__(256) void pack_weights(
    const float* __restrict__ w, const float* __restrict__ g,
    const float* __restrict__ v, unsigned short* __restrict__ dst,
    int I, int KHW, int kcTot)
{
  int i = blockIdx.x * 256 + threadIdx.x;
  int total = kcTot * 3072;
  if (i >= total) return;
  int e    = i & 15;
  int lane = (i >> 4) & 31;
  int mt   = (i >> 9) % 6;
  int kc   = i / 3072;
  int m = mt * 16 + (lane & 15);
  int K = kc * 32 + ((lane & 16) ? 16 : 0) + e;
  float val = 0.f;
  if (K < I * KHW) {
    int ci = K % I, tap = K / I;
    val = w[(m * I + ci) * KHW + tap];
    if (g) val *= g[m] * rsqrtf(v[m] + 1e-5f);
  }
  dst[i] = f2bf(val);
}

__global__ void fuse_bias(const float* __restrict__ cb, const float* __restrict__ g,
                          const float* __restrict__ be, const float* __restrict__ m,
                          const float* __restrict__ v, float* __restrict__ out)
{
  int i = threadIdx.x;
  if (i < 96) {
    float s = g[i] * rsqrtf(v[i] + 1e-5f);
    out[i] = (cb[i] - m[i]) * s + be[i];
  }
}

// ---------------------------------------------------------------------------
// 3x3 conv + folded BN + ReLU as implicit GEMM with bf16 WMMA.
// One workgroup = one (b,h) row. LDS holds x[b,:,h-1..h+1,:] as bf16 laid out
// [kh][w+1 (66 cols, zero-padded borders)][ci]. Fully unrolled K loop with all
// fragment addresses as base + constant immediate so loads can be clause-
// batched ahead of the WMMAs and fill the WMMA->WMMA hazard slots.
// ---------------------------------------------------------------------------
__global__ __launch_bounds__(256) void conv3_bn_relu_wmma(
    const float* __restrict__ in, const unsigned short* __restrict__ pw,
    const float* __restrict__ bias, float* __restrict__ out)
{
  __shared__ __attribute__((aligned(32))) unsigned short xs[3 * 66 * 96];
  int bh = blockIdx.x;
  int b = bh >> 6, h = bh & 63;
  int tid = threadIdx.x;
  // interior cells: ww_store = w+1 in [1,64]
  for (int idx = tid; idx < 96 * 3 * 64; idx += 256) {
    int ci = idx / 192, rem = idx % 192;
    int kh = rem >> 6, w = rem & 63;
    int hh = h + kh - 1;
    float val = 0.f;
    if ((unsigned)hh < 64u) val = in[((b * 96 + ci) * 64 + hh) * 64 + w];
    xs[(kh * 66 + w + 1) * 96 + ci] = f2bf(val);
  }
  // zero the border columns ww_store = 0 and 65
  for (int idx = tid; idx < 3 * 2 * 96; idx += 256) {
    int kh = idx / 192;
    int border = (idx >> 6) & 1;       // 0 -> col 0, 1 -> col 65
    int ci = idx % 96;
    xs[(kh * 66 + border * 65) * 96 + ci] = 0;
  }
  __syncthreads();

  int lane = tid & 31, wid = tid >> 5;
  int nt = wid & 3, mg = wid >> 2;
  int w0 = nt * 16;
  int n = lane & 15;
  int khalf = (lane & 16) ? 16 : 0;

  // single base registers; everything else is a compile-time offset
  const unsigned short* abase = pw + (size_t)(mg * 3 * 32 + lane) * 16;
  const unsigned short* bbase = xs + (w0 + n) * 96 + khalf;

  v8f acc[3] = {};
#pragma unroll
  for (int kh = 0; kh < 3; ++kh) {
#pragma unroll
    for (int kw = 0; kw < 3; ++kw) {
#pragma unroll
      for (int cb = 0; cb < 3; ++cb) {
        const int kc = (kh * 3 + kw) * 3 + cb;
        v16bf bf = __builtin_bit_cast(
            v16bf, *(const v16u*)(bbase + (kh * 66 + kw) * 96 + cb * 32));
#pragma unroll
        for (int i = 0; i < 3; ++i) {
          v16bf af = __builtin_bit_cast(
              v16bf, *(const v16u*)(abase + kc * 3072 + i * 512));
          acc[i] = __builtin_amdgcn_wmma_f32_16x16x32_bf16(
              false, af, false, bf, (short)0, acc[i], false, false);
        }
      }
    }
  }
  int rowHi = (lane & 16) ? 8 : 0;
#pragma unroll
  for (int i = 0; i < 3; ++i) {
    int mbase = (mg * 3 + i) * 16;
#pragma unroll
    for (int r = 0; r < 8; ++r) {
      int m = mbase + r + rowHi;
      float val = acc[i][r] + bias[m];
      val = val > 0.f ? val : 0.f;
      out[((b * 96 + m) * 64 + h) * 64 + w0 + n] = val;
    }
  }
}

// ---------------------------------------------------------------------------
// S6 phase 0: gather path-ordered x into LDS (and write xPath), compute
// proj = x@W_x (38 cols), split dt/B/C, delta = softplus(dt@W_dt + b_dt).
// Layouts: xPath/delta [dir][b][d][t], Bm/Cm [dir][b][t][n].
// ---------------------------------------------------------------------------
__global__ __launch_bounds__(256) void s6_proj(
    const float* __restrict__ y2, const float* __restrict__ Wx,
    const float* __restrict__ Wdt, const float* __restrict__ bdt,
    float* __restrict__ xPath, float* __restrict__ delta,
    float* __restrict__ Bm, float* __restrict__ Cm)
{
  __shared__ float xsF[96 * 64];
  __shared__ float pr[64 * 8];
  int g = blockIdx.x;
  int dir = g >> 8, b = (g >> 6) & 3, tb = g & 63;
  int dirb = dir * 4 + b;
  int tid = threadIdx.x;

  for (int idx = tid; idx < 6144; idx += 256) {
    int d = idx >> 6, tt = idx & 63;
    int t = tb * 64 + tt;
    int q = t >> 6, rem = t & 63;
    int hh, wp;
    if (dir == 0)      { hh = q;        wp = rem;      }
    else if (dir == 1) { hh = q;        wp = 63 - rem; }
    else if (dir == 2) { hh = rem;      wp = q;        }
    else               { hh = 63 - rem; wp = q;        }
    float val = y2[((b * 96 + d) * 64 + hh) * 64 + wp];
    xsF[idx] = val;
    xPath[(((size_t)dirb * 96 + d) << 12) + t] = val;
  }
  __syncthreads();

  for (int idx = tid; idx < 64 * 38; idx += 256) {
    int tt = idx / 38, col = idx % 38;
    const float* wc = Wx + dir * 96 * 38 + col;
    float s = 0.f;
    for (int d = 0; d < 96; ++d) s += xsF[d * 64 + tt] * wc[d * 38];
    int t = tb * 64 + tt;
    if (col < 6)       pr[tt * 8 + col] = s;
    else if (col < 22) Bm[((((size_t)dirb << 12) + t) << 4) + (col - 6)]  = s;
    else               Cm[((((size_t)dirb << 12) + t) << 4) + (col - 22)] = s;
  }
  __syncthreads();

  for (int idx = tid; idx < 6144; idx += 256) {
    int d = idx >> 6, tt = idx & 63;
    float s = bdt[dir * 96 + d];
#pragma unroll
    for (int r = 0; r < 6; ++r) s += pr[tt * 8 + r] * Wdt[(dir * 6 + r) * 96 + d];
    float sp = (s > 20.f) ? s : log1pf(__expf(s));
    delta[(((size_t)dirb * 96 + d) << 12) + tb * 64 + tt] = sp;
  }
}

// ---------------------------------------------------------------------------
// S6 phase 1: per (dir,b,chunk) local scan from h=0. 512 threads x 3 states,
// tt-outer so the 3 independent exp/fma recurrence chains interleave.
// ---------------------------------------------------------------------------
__global__ __launch_bounds__(512) void s6_chunk(
    const float* __restrict__ xPath, const float* __restrict__ delta,
    const float* __restrict__ Bm, const float* __restrict__ Alog,
    float* __restrict__ localH, float* __restrict__ prodA)
{
  int g = blockIdx.x;                 // (dir*4+b)*64 + c
  int dirb = g >> 6, c = g & 63;
  int dir = dirb >> 2;
  int tid = threadIdx.x;

  float Ac[3], h[3], p[3];
  const float *xr[3], *dr[3], *br[3];
#pragma unroll
  for (int i = 0; i < 3; ++i) {
    int s = tid + i * 512;
    int d = s >> 4, n = s & 15;
    Ac[i] = -__expf(Alog[(dir * 96 + d) * 16 + n]);
    xr[i] = xPath + (((size_t)dirb * 96 + d) << 12) + c * 64;
    dr[i] = delta + (((size_t)dirb * 96 + d) << 12) + c * 64;
    br[i] = Bm + ((((size_t)dirb << 12) + c * 64) << 4) + n;
    h[i] = 0.f; p[i] = 1.f;
  }
  for (int tt = 0; tt < 64; ++tt) {
#pragma unroll
    for (int i = 0; i < 3; ++i) {
      float dl = dr[i][tt];
      float a = __expf(dl * Ac[i]);
      h[i] = a * h[i] + dl * br[i][tt * 16] * xr[i][tt];
      p[i] *= a;
    }
  }
#pragma unroll
  for (int i = 0; i < 3; ++i) {
    size_t o = (size_t)g * 1536 + tid + i * 512;
    localH[o] = h[i];
    prodA[o]  = p[i];
  }
}

// ---------------------------------------------------------------------------
// S6 phase 2: serial carry scan over the 64 chunks per (dir,b,d,n).
// ---------------------------------------------------------------------------
__global__ __launch_bounds__(256) void s6_carry(
    const float* __restrict__ localH, const float* __restrict__ prodA,
    float* __restrict__ carryIn)
{
  int gid = blockIdx.x * 256 + threadIdx.x;   // < 24576
  int dirb = gid / 1536, s = gid % 1536;
  float carry = 0.f;
  for (int c = 0; c < 64; ++c) {
    size_t o = ((size_t)dirb * 64 + c) * 1536 + s;
    carryIn[o] = carry;
    carry = prodA[o] * carry + localH[o];
  }
}

// ---------------------------------------------------------------------------
// S6 phase 3: replay chunk with carried-in state; y_d = sum_n h*C_n + D_d*x_d
// reduced over the 16 n-lanes with width-16 xor-shuffles (wave32). tt-outer,
// 3 interleaved chains.
// ---------------------------------------------------------------------------
__global__ __launch_bounds__(512) void s6_final(
    const float* __restrict__ xPath, const float* __restrict__ delta,
    const float* __restrict__ Bm, const float* __restrict__ Cm,
    const float* __restrict__ Alog, const float* __restrict__ Dskip,
    const float* __restrict__ carryIn, float* __restrict__ ydir)
{
  int g = blockIdx.x;
  int dirb = g >> 6, c = g & 63;
  int dir = dirb >> 2;
  int tid = threadIdx.x;
  int n0 = tid & 15;

  float Ac[3], Dd[3], h[3];
  const float *xr[3], *dr[3], *br[3], *cr[3];
  float *yo[3];
#pragma unroll
  for (int i = 0; i < 3; ++i) {
    int s = tid + i * 512;
    int d = s >> 4, n = s & 15;
    Ac[i] = -__expf(Alog[(dir * 96 + d) * 16 + n]);
    Dd[i] = Dskip[dir * 96 + d];
    xr[i] = xPath + (((size_t)dirb * 96 + d) << 12) + c * 64;
    dr[i] = delta + (((size_t)dirb * 96 + d) << 12) + c * 64;
    br[i] = Bm + ((((size_t)dirb << 12) + c * 64) << 4) + n;
    cr[i] = Cm + ((((size_t)dirb << 12) + c * 64) << 4) + n;
    yo[i] = ydir + (((size_t)dirb * 96 + d) << 12) + c * 64;
    h[i] = carryIn[(size_t)g * 1536 + s];
  }
  for (int tt = 0; tt < 64; ++tt) {
    float contrib[3];
#pragma unroll
    for (int i = 0; i < 3; ++i) {
      float xv = xr[i][tt];
      float dl = dr[i][tt];
      float a = __expf(dl * Ac[i]);
      h[i] = a * h[i] + dl * br[i][tt * 16] * xv;
      contrib[i] = h[i] * cr[i][tt * 16];
    }
#pragma unroll
    for (int m = 8; m >= 1; m >>= 1) {
#pragma unroll
      for (int i = 0; i < 3; ++i) contrib[i] += __shfl_xor(contrib[i], m, 16);
    }
    if (n0 == 0) {
#pragma unroll
      for (int i = 0; i < 3; ++i) yo[i][tt] = contrib[i] + Dd[i] * xr[i][tt];
    }
  }
}

// ---------------------------------------------------------------------------
// Mean over 4 directions (no un-flip, per reference) + 1x1 conv via bf16 WMMA.
// comb tile stored [t][ci] in LDS -> contiguous 32B B-fragment loads.
// ---------------------------------------------------------------------------
__global__ __launch_bounds__(256) void adj_wmma(
    const float* __restrict__ ydir, const unsigned short* __restrict__ pw,
    const float* __restrict__ adjb, float* __restrict__ out)
{
  __shared__ __attribute__((aligned(32))) unsigned short cs[64 * 96];  // [tt][ci]
  int bt = blockIdx.x;
  int b = bt >> 6, tb = bt & 63;
  int tid = threadIdx.x;
  for (int idx = tid; idx < 96 * 64; idx += 256) {
    int d = idx >> 6, tt = idx & 63;
    size_t t = (size_t)tb * 64 + tt;
    float s = 0.f;
#pragma unroll
    for (int dir = 0; dir < 4; ++dir)
      s += ydir[(((size_t)(dir * 4 + b) * 96 + d) << 12) + t];
    cs[tt * 96 + d] = f2bf(0.25f * s);
  }
  __syncthreads();

  int lane = tid & 31, wid = tid >> 5;
  int nt = wid & 3, mg = wid >> 2;
  int n = lane & 15;
  int khalf = (lane & 16) ? 16 : 0;

  const unsigned short* abase = pw + (size_t)(mg * 3 * 32 + lane) * 16;
  const unsigned short* bbase = cs + (nt * 16 + n) * 96 + khalf;

  v8f acc[3] = {};
#pragma unroll
  for (int kc = 0; kc < 3; ++kc) {
    v16bf bf = __builtin_bit_cast(v16bf, *(const v16u*)(bbase + kc * 32));
#pragma unroll
    for (int i = 0; i < 3; ++i) {
      v16bf af = __builtin_bit_cast(v16bf, *(const v16u*)(abase + kc * 3072 + i * 512));
      acc[i] = __builtin_amdgcn_wmma_f32_16x16x32_bf16(
          false, af, false, bf, (short)0, acc[i], false, false);
    }
  }
  int rowHi = (lane & 16) ? 8 : 0;
#pragma unroll
  for (int i = 0; i < 3; ++i) {
    int mbase = (mg * 3 + i) * 16;
#pragma unroll
    for (int r = 0; r < 8; ++r) {
      int m = mbase + r + rowHi;
      out[(((size_t)b * 96 + m) << 12) + tb * 64 + nt * 16 + n] = acc[i][r] + adjb[m];
    }
  }
}

// ---------------------------------------------------------------------------
extern "C" void kernel_launch(void* const* d_in, const int* in_sizes, int n_in,
                              void* d_out, int out_size, void* d_ws, size_t ws_size,
                              hipStream_t stream) {
  const float* x     = (const float*)d_in[0];
  const float* c1w   = (const float*)d_in[1];
  const float* c1b   = (const float*)d_in[2];
  const float* bn1g  = (const float*)d_in[3];
  const float* bn1b  = (const float*)d_in[4];
  const float* bn1m  = (const float*)d_in[5];
  const float* bn1v  = (const float*)d_in[6];
  const float* c2w   = (const float*)d_in[7];
  const float* c2b   = (const float*)d_in[8];
  const float* bn2g  = (const float*)d_in[9];
  const float* bn2b  = (const float*)d_in[10];
  const float* bn2m  = (const float*)d_in[11];
  const float* bn2v  = (const float*)d_in[12];
  const float* Alog  = (const float*)d_in[13];
  const float* Dsk   = (const float*)d_in[14];
  const float* Wx    = (const float*)d_in[15];
  const float* Wdt   = (const float*)d_in[16];
  const float* bdt   = (const float*)d_in[17];
  const float* adjw  = (const float*)d_in[18];
  const float* adjb  = (const float*)d_in[19];
  float* out = (float*)d_out;

  char* ws = (char*)d_ws;
  size_t off = 0;
  auto alloc = [&](size_t bytes) {
    char* p = ws + off;
    off = (off + bytes + 255) & ~(size_t)255;
    return (void*)p;
  };
  unsigned short* pw1 = (unsigned short*)alloc(82944 * 2);
  unsigned short* pw2 = (unsigned short*)alloc(82944 * 2);
  unsigned short* pwa = (unsigned short*)alloc(9216 * 2);
  float* bias1   = (float*)alloc(96 * 4);
  float* bias2   = (float*)alloc(96 * 4);
  float* y1      = (float*)alloc((size_t)1572864 * 4);
  float* y2      = (float*)alloc((size_t)1572864 * 4);
  float* xPath   = (float*)alloc((size_t)6291456 * 4);
  float* delta   = (float*)alloc((size_t)6291456 * 4);
  float* Bm      = (float*)alloc((size_t)1048576 * 4);
  float* Cm      = (float*)alloc((size_t)1048576 * 4);
  float* localH  = (float*)alloc((size_t)1572864 * 4);
  float* prodA   = (float*)alloc((size_t)1572864 * 4);
  float* carryIn = (float*)alloc((size_t)1572864 * 4);
  float* ydir    = (float*)alloc((size_t)6291456 * 4);

  pack_weights<<<324, 256, 0, stream>>>(c1w, bn1g, bn1v, pw1, 96, 9, 27);
  pack_weights<<<324, 256, 0, stream>>>(c2w, bn2g, bn2v, pw2, 96, 9, 27);
  pack_weights<<<36, 256, 0, stream>>>(adjw, nullptr, nullptr, pwa, 96, 1, 3);
  fuse_bias<<<1, 96, 0, stream>>>(c1b, bn1g, bn1b, bn1m, bn1v, bias1);
  fuse_bias<<<1, 96, 0, stream>>>(c2b, bn2g, bn2b, bn2m, bn2v, bias2);

  conv3_bn_relu_wmma<<<256, 256, 0, stream>>>(x, pw1, bias1, y1);
  conv3_bn_relu_wmma<<<256, 256, 0, stream>>>(y1, pw2, bias2, y2);

  s6_proj<<<1024, 256, 0, stream>>>(y2, Wx, Wdt, bdt, xPath, delta, Bm, Cm);
  s6_chunk<<<1024, 512, 0, stream>>>(xPath, delta, Bm, Alog, localH, prodA);
  s6_carry<<<96, 256, 0, stream>>>(localH, prodA, carryIn);
  s6_final<<<1024, 512, 0, stream>>>(xPath, delta, Bm, Cm, Alog, Dsk, carryIn, ydir);

  adj_wmma<<<256, 256, 0, stream>>>(ydir, pwa, adjb, out);
}
